// FreqLinear_35424890257670
// MI455X (gfx1250) — matrix operand order repaired and checked by reading, not compiled, not run on hardware
//
#include <hip/hip_runtime.h>
#include <cmath>

#define IN_F   4096
#define OUT_F  8192
#define MROWS  256
#define NLEVM1 255

#define BM 128
#define BN 128
#define BK 32
#define LDSP 40         // padded row stride (bf16): 80B, 16B-aligned, conflict-free frags
#define NCHUNK (IN_F / BK)   // 128

typedef __attribute__((ext_vector_type(16))) __bf16 bf16x16;
typedef __attribute__((ext_vector_type(8)))  float  f32x8;
typedef __attribute__((ext_vector_type(4)))  unsigned int u32x4;
typedef __attribute__((ext_vector_type(2)))  unsigned int u32x2;

struct Basis { float b[8][8]; };    // DCT-II synthesis basis, t=0..7 half (mirror symmetry)

union FragU  { bf16x16 v; u32x4 q[2]; };
union Pack16 { __bf16 h[16]; u32x4 q[2]; };
union Pack4  { __bf16 h[4];  u32x2 q; };

__global__ __launch_bounds__(256) void freq_linear_wmma(
    const float* __restrict__ x,        // [256][4096] f32
    const int*   __restrict__ idx,      // [OUT_F*IN_F/16][8] int32
    const float* __restrict__ c_min,    // [8]
    const float* __restrict__ c_range,  // [8]
    const float* __restrict__ bias,     // [8192]
    float*       __restrict__ out,      // [256][8192]
    const Basis  bas)
{
  __shared__ __bf16 sA[2][BM * LDSP];   // x tile   bf16 [m][k]  2 x 10 KB
  __shared__ __bf16 sB[2][BN * LDSP];   // W tile   bf16 [n][k]  2 x 10 KB

  const int tid  = threadIdx.x;
  const int lane = tid & 31;
  const int wave = tid >> 5;            // 0..7
  const int wm   = wave >> 2;           // 0..1 : 64 M-rows per wave
  const int wn   = wave & 3;            // 0..3 : 32 N-cols per wave
  const int mtile = blockIdx.y * BM;
  const int ntile = blockIdx.x * BN;

  // per-frequency dequant constants (uniform -> scalar path)
  float scl[8], mn[8];
#pragma unroll
  for (int k = 0; k < 8; ++k) {
    scl[k] = c_range[k] * (1.0f / NLEVM1);
    mn[k]  = c_min[k];
  }

  f32x8 acc[4][2] = {};

  // uniform per-chunk work: every thread does 1 dequant block + 4 float4 of x
  const int drow = tid >> 1;            // 0..127 (N row)
  const int jb   = tid & 1;             // K block in chunk
  const int xr   = tid >> 3;            // 0..31  (x row helper)
  const int xc4  = (tid & 7) << 2;      // 0..28  (x col)

  // WMMA fragment lane decode (16-bit A 16x32 / B 32x16, wave32; contiguous k = 16*(lane/16) + s in A and B)
  const int am  = lane & 15;
  const int akb = (lane >> 4) << 4;     // A K base per lane-half (0/16)
  const int bn  = lane & 15;
  const int bkb = (lane >> 4) << 4;     // B K base per lane-half (0/16)

  // ---- register staging: global loads issued one chunk ahead ----
  float4 xs[4];
  int4 q0, q1;
  auto load_global = [&](int kk) {
    const int blk = (ntile + drow) * (IN_F / 16) + (kk >> 4) + jb;
    q0 = *reinterpret_cast<const int4*>(idx + (size_t)blk * 8);
    q1 = *reinterpret_cast<const int4*>(idx + (size_t)blk * 8 + 4);
#pragma unroll
    for (int it = 0; it < 4; ++it) {
      const int row = it * 32 + xr;
      xs[it] = *reinterpret_cast<const float4*>(
          x + (size_t)(mtile + row) * IN_F + kk + xc4);
    }
  };

  auto store_stage = [&](int buf) {
    // dequant + symmetric IDCT -> sB
    const float c0  = (float)q0.x * scl[0] + mn[0];
    const float c1  = (float)q0.y * scl[1] + mn[1];
    const float c2  = (float)q0.z * scl[2] + mn[2];
    const float c3  = (float)q0.w * scl[3] + mn[3];
    const float c4f = (float)q1.x * scl[4] + mn[4];
    const float c5  = (float)q1.y * scl[5] + mn[5];
    const float c6  = (float)q1.z * scl[6] + mn[6];
    const float c7  = (float)q1.w * scl[7] + mn[7];
    Pack16 w;
    // B[k][15-t] = (-1)^k B[k][t]: even/odd split halves the FMA count
#pragma unroll
    for (int t = 0; t < 8; ++t) {
      float e = c0 * bas.b[0][t];
      e = fmaf(c2,  bas.b[2][t], e);
      e = fmaf(c4f, bas.b[4][t], e);
      e = fmaf(c6,  bas.b[6][t], e);
      float o = c1 * bas.b[1][t];
      o = fmaf(c3,  bas.b[3][t], o);
      o = fmaf(c5,  bas.b[5][t], o);
      o = fmaf(c7,  bas.b[7][t], o);
      w.h[t]      = (__bf16)(e + o);
      w.h[15 - t] = (__bf16)(e - o);
    }
    u32x4* dst = reinterpret_cast<u32x4*>(&sB[buf][drow * LDSP + jb * 16]);
    dst[0] = w.q[0];
    dst[1] = w.q[1];
    // x tile cvt -> sA
#pragma unroll
    for (int it = 0; it < 4; ++it) {
      const int row = it * 32 + xr;
      Pack4 p;
      p.h[0] = (__bf16)xs[it].x; p.h[1] = (__bf16)xs[it].y;
      p.h[2] = (__bf16)xs[it].z; p.h[3] = (__bf16)xs[it].w;
      *reinterpret_cast<u32x2*>(&sA[buf][row * LDSP + xc4]) = p.q;
    }
  };

  auto wmma_stage = [&](int buf) {
    FragU bfr[2];
#pragma unroll
    for (int j = 0; j < 2; ++j) {
      const __bf16* p = &sB[buf][(size_t)(wn * 32 + j * 16 + bn) * LDSP + bkb];
      bfr[j].q[0] = *reinterpret_cast<const u32x4*>(p);
      bfr[j].q[1] = *reinterpret_cast<const u32x4*>(p + 8);
    }
    FragU afr[4];
#pragma unroll
    for (int i = 0; i < 4; ++i) {
      const __bf16* p = &sA[buf][(size_t)(wm * 64 + i * 16 + am) * LDSP + akb];
      afr[i].q[0] = *reinterpret_cast<const u32x4*>(p);
      afr[i].q[1] = *reinterpret_cast<const u32x4*>(p + 8);
    }
#pragma unroll
    for (int i = 0; i < 4; ++i)
#pragma unroll
      for (int j = 0; j < 2; ++j)
        acc[i][j] = __builtin_amdgcn_wmma_f32_16x16x32_bf16(
            false, afr[i].v, false, bfr[j].v, (short)0, acc[i][j],
            false, false);
  };

  // ---- prologue ----
  load_global(0);
  store_stage(0);
  __syncthreads();

  // ---- main loop: 2 chunks per iteration, static buffer indices, 1 barrier/chunk.
  // dequant/cvt VALU of chunk i+1 co-executes with WMMAs of chunk i. ----
#pragma unroll 1
  for (int ch = 0; ch < NCHUNK; ch += 2) {
    load_global((ch + 1) * BK);          // ch+1 <= 127: always valid
    wmma_stage(0);
    store_stage(1);
    __syncthreads();

    if (ch + 2 < NCHUNK) load_global((ch + 2) * BK);
    wmma_stage(1);
    if (ch + 2 < NCHUNK) store_stage(0);
    __syncthreads();
  }

  // ---- epilogue: bias add + store (C/D layout: VGPR r -> M = r + 8*(lane>=16)) ----
  const int mhi = (lane >> 4) * 8;
#pragma unroll
  for (int j = 0; j < 2; ++j) {
    const int col = ntile + wn * 32 + j * 16 + (lane & 15);
    const float bv = bias[col];
#pragma unroll
    for (int i = 0; i < 4; ++i) {
      const int m0 = mtile + wm * 64 + i * 16 + mhi;
#pragma unroll
      for (int r = 0; r < 8; ++r)
        out[(size_t)(m0 + r) * OUT_F + col] = acc[i][j][r] + bv;
    }
  }
}

extern "C" void kernel_launch(void* const* d_in, const int* in_sizes, int n_in,
                              void* d_out, int out_size, void* d_ws, size_t ws_size,
                              hipStream_t stream) {
  (void)in_sizes; (void)n_in; (void)out_size; (void)d_ws; (void)ws_size;
  const float* x       = (const float*)d_in[0];
  const int*   idx     = (const int*)d_in[1];
  const float* c_min   = (const float*)d_in[2];
  const float* c_range = (const float*)d_in[3];
  const float* bias    = (const float*)d_in[4];
  float* out = (float*)d_out;

  // Host-side DCT-II synthesis basis, half table (t = 0..7); wave-uniform -> kernarg
  Basis bas;
  const double pi = 3.14159265358979323846;
  for (int k = 0; k < 8; ++k)
    for (int t = 0; t < 8; ++t) {
      double b = std::cos(pi * (t + 0.5) * k / 16.0) * std::sqrt(2.0 / 16.0);
      if (k == 0) b *= 1.0 / std::sqrt(2.0);
      bas.b[k][t] = (float)b;
    }

  dim3 grid(OUT_F / BN, MROWS / BM);   // 64 x 2 = 128 workgroups -> 128 WGPs engaged
  dim3 block(256);                     // 8 waves (wave32)
  freq_linear_wmma<<<grid, block, 0, stream>>>(x, idx, c_min, c_range, bias, out, bas);
}
